// VectorExplorer_32358283608385
// MI455X (gfx1250) — matrix operand order repaired and checked by reading, not compiled, not run on hardware
//
#include <hip/hip_runtime.h>
#include <hip/hip_bf16.h>

typedef __attribute__((ext_vector_type(16))) _Float16 v16h;
typedef __attribute__((ext_vector_type(8)))  float    v8f;
typedef __attribute__((ext_vector_type(4)))  float    v4f;

#define BB 32
#define DD 32
#define TT 16384
#define NN 128
#define NTILES 8            // NN/16
#define WAVES 4
#define TILES_PER_WAVE 8
#define T_PER_BLOCK (WAVES * TILES_PER_WAVE * 16)   // 512
#define NPAD (NN + 4)       // sims row stride (floats): 132 = 33*4 -> 16B-aligned rows, conflict-free b128
#define DPAD (DD + 1)       // ref row stride

__global__ __launch_bounds__(128)
void VectorExplorer_topk_wmma(const float* __restrict__ src,
                              const float* __restrict__ tok,
                              float* __restrict__ out) {
    __shared__ float    ref_raw[NN][DPAD];                 // raw tokens, row-major (n, d)
    __shared__ _Float16 bfrag[NTILES * 32 * 16];           // WMMA B fragments, per-lane layout
    __shared__ float    sims[WAVES][16][NPAD];             // per-wave 16 x 128 similarity slab

    const int tid  = threadIdx.x;
    const int lane = tid & 31;
    const int wave = tid >> 5;
    const int m    = lane & 15;      // row within 16-row tile
    const int hi   = lane >> 4;      // lane half

    // ---- Phase 0: token preprocessing (one thread per token n) ----
    {
        const int n = tid;           // blockDim.x == NN == 128
        float v[DD];
        float ss = 0.0f;
        #pragma unroll
        for (int d = 0; d < DD; ++d) {
            float x = tok[d * NN + n];
            v[d] = x;
            ref_raw[n][d] = x;
            ss += x * x;
        }
        const float inv = __frsqrt_rn(ss);
        const int ntile = n >> 4, nl = n & 15;
        // Scatter normalized f16 token into this kernel's contiguous-k B-fragment layout:
        // lanes 0-15 hold K=0..15 (half j=K), lanes 16-31 hold K=16..31 (half j=K-16)
        #pragma unroll
        for (int k = 0; k < DD; ++k) {
            const int ln = nl + ((k >> 4) << 4);
            const int j  = k & 15;
            bfrag[(ntile * 32 + ln) * 16 + j] = (_Float16)(v[k] * inv);
        }
    }
    __syncthreads();

    const int b    = blockIdx.x / (TT / T_PER_BLOCK);
    const int tblk = blockIdx.x % (TT / T_PER_BLOCK);
    const float* __restrict__ sb = src + (size_t)b * DD * TT;
    float* __restrict__       ob = out + (size_t)b * DD * TT;

    const v16h* __restrict__ bf = (const v16h*)bfrag;

    for (int tt = 0; tt < TILES_PER_WAVE; ++tt) {
        const int t0 = tblk * T_PER_BLOCK + (wave * TILES_PER_WAVE + tt) * 16;

        // ---- A fragment: 16 t-rows x 32 D, f32 -> f16, contiguous-k order (same as bfrag) ----
        // half j (lane lo): K=j (0..15) ; lane hi: K=16+j
        v16h a;
        #pragma unroll
        for (int j = 0; j < 16; ++j) {
            const int k = j + 16 * hi;
            a[j] = (_Float16)sb[(size_t)k * TT + t0 + m];
        }

        // ---- 8 WMMAs: 16x16 similarity tiles, f32 accumulate, spill to LDS ----
        #pragma unroll
        for (int nt = 0; nt < NTILES; ++nt) {
            const v16h bv = bf[nt * 32 + lane];
            v8f c = {};
            c = __builtin_amdgcn_wmma_f32_16x16x32_f16(
                    /*neg_a=*/false, a, /*neg_b=*/false, bv,
                    /*c_mod=*/(short)0, c, /*reuse_a=*/false, /*reuse_b=*/false);
            #pragma unroll
            for (int v = 0; v < 8; ++v) {
                // lane<16: row v, col lane ; lane>=16: row v+8, col lane-16
                sims[wave][v + 8 * hi][nt * 16 + m] = c[v];
            }
        }

        // Same-wave cross-lane LDS dependency: wait for DS stores to land.
        asm volatile("s_wait_dscnt 0x0" ::: "memory");

        // ---- Top-4 one-pass insertion over 128 sims, vectorized: 32 x ds_load_b128 ----
        // Lanes l and l+16 scan the same row (LDS same-address broadcast), so both
        // halves own the top-4 indices for the d-split output phase.
        float v0 = -1e30f, v1 = -1e30f, v2 = -1e30f, v3 = -1e30f;
        int   i0 = 0, i1 = 0, i2 = 0, i3 = 0;
        const v4f* __restrict__ row = (const v4f*)&sims[wave][m][0];
        #pragma unroll 4
        for (int q = 0; q < NN / 4; ++q) {
            const v4f x4 = row[q];
            #pragma unroll
            for (int c = 0; c < 4; ++c) {
                const float x = x4[c];
                const int   i = q * 4 + c;
                if (x > v3) {
                    if (x > v0)      { v3=v2; i3=i2; v2=v1; i2=i1; v1=v0; i1=i0; v0=x; i0=i; }
                    else if (x > v1) { v3=v2; i3=i2; v2=v1; i2=i1; v1=x;  i1=i; }
                    else if (x > v2) { v3=v2; i3=i2; v2=x;  i2=i; }
                    else             { v3=x;  i3=i; }
                }
            }
        }

        // ---- Gather + mean of 4 raw token rows; d range split across lane halves ----
        #pragma unroll
        for (int dd = 0; dd < 16; ++dd) {
            const int d = hi * 16 + dd;
            const float r = 0.25f * (ref_raw[i0][d] + ref_raw[i1][d] +
                                     ref_raw[i2][d] + ref_raw[i3][d]);
            ob[(size_t)d * TT + t0 + m] = r;
        }
    }
}

extern "C" void kernel_launch(void* const* d_in, const int* in_sizes, int n_in,
                              void* d_out, int out_size, void* d_ws, size_t ws_size,
                              hipStream_t stream) {
    const float* src = (const float*)d_in[0];   // (B, D, T) f32
    const float* tok = (const float*)d_in[1];   // (1, D, N) f32
    float* outp = (float*)d_out;                // (B, D, T) f32

    const int blocks = BB * (TT / T_PER_BLOCK); // 32 * 32 = 1024
    VectorExplorer_topk_wmma<<<blocks, 128, 0, stream>>>(src, tok, outp);
}
